// EnhancedSemGCN_83880711291028
// MI455X (gfx1250) — matrix operand order repaired and checked by reading, not compiled, not run on hardware
//
#include <hip/hip_runtime.h>
#include <stdint.h>

// ---------------------------------------------------------------------------
// EnhancedSemGCN for MI455X (gfx1250, wave32, WMMA).
// All matmuls -> v_wmma_f32_16x16x32_f16 (f16 mul, f32 acc).
// B operands pre-converted to f16 (N,K) so hot-loop staging is pure b128
// copies; A staged with float4 loads + packed f16x2 stores.
// ---------------------------------------------------------------------------

typedef __attribute__((ext_vector_type(16))) _Float16 v16h;
typedef __attribute__((ext_vector_type(8)))  float    v8f;

union FragH { v16h h; uint32_t u[8]; };

__device__ __forceinline__ uint32_t pack_f16x2(float a, float b) {
  union { _Float16 h[2]; uint32_t u; } p;
  p.h[0] = (_Float16)a; p.h[1] = (_Float16)b;
  return p.u;
}

// ---------------------------------------------------------------------------
// Strided-batched WMMA GEMM:
//   C[z] = act( ((A[z] @ B[z]^T) * scale + bias) / divisor[row] )
// A: (M,K) f32 row-major.  Bg: (N,K) f16 row-major (pre-transposed weight /
// activation).  C: (M,N) f32 (+ optional f16 copy C16).
// ALIGNED: M%64==0, N%64==0, K%32==0 -> guard-free vectorized staging.
// Block = 128 threads (4 waves); tile 64x64; wave 32x32 (2x2 16x16 frags).
// ---------------------------------------------------------------------------
template <bool ALIGNED>
__global__ __launch_bounds__(128)
void semgcn_gemm(const float* __restrict__ A, long sA, int lda,
                 const _Float16* __restrict__ Bg, long sB, int ldb,
                 float* __restrict__ C, long sC, int ldc,
                 _Float16* __restrict__ C16,
                 const float* __restrict__ bias,
                 const float* __restrict__ divisor,
                 float scale, int act, int M, int N, int K)
{
  const int tid  = threadIdx.x;
  const int wave = tid >> 5;
  const int lane = tid & 31;
  const int l16  = lane & 15;
  const int lh   = lane >> 4;
  const int wr   = wave >> 1;
  const int wc   = wave & 1;
  const int m0   = blockIdx.y << 6;
  const int n0   = blockIdx.x << 6;
  const float*    Ab  = A  + (long)blockIdx.z * sA;
  const _Float16* Bb  = Bg + (long)blockIdx.z * sB;
  float*          Cb  = C  + (long)blockIdx.z * sC;

  __shared__ _Float16 Ah[64 * 40];   // A tile 64x32 f16, stride 40
  __shared__ _Float16 Bt[64 * 40];   // B tile (n,k) f16, stride 40

  v8f acc[2][2];
#pragma unroll
  for (int i = 0; i < 2; ++i)
#pragma unroll
    for (int j = 0; j < 2; ++j) { v8f z = {}; acc[i][j] = z; }

  for (int k0 = 0; k0 < K; k0 += 32) {
    __syncthreads();
    if constexpr (ALIGNED) {
      // A: 512 float4 loads, convert, packed uint2 stores
#pragma unroll
      for (int i = 0; i < 4; ++i) {
        int idx = tid + (i << 7);          // 0..511
        int r = idx >> 3;
        int c = (idx & 7) << 2;
        const float4 f = *(const float4*)(Ab + (long)(m0 + r) * lda + k0 + c);
        uint2 u;
        u.x = pack_f16x2(f.x, f.y);
        u.y = pack_f16x2(f.z, f.w);
        *(uint2*)&Ah[r * 40 + c] = u;
      }
      // B: raw f16, 256 uint4 copies (global_load_b128 -> ds_store_b128)
#pragma unroll
      for (int i = 0; i < 2; ++i) {
        int idx = tid + (i << 7);          // 0..255
        int r = idx >> 2;
        int c = (idx & 3) << 3;
        const uint4 u = *(const uint4*)(Bb + (long)(n0 + r) * ldb + k0 + c);
        *(uint4*)&Bt[r * 40 + c] = u;
      }
      if (k0 + 32 < K)
        __builtin_prefetch(Ab + (long)(m0 + (tid & 63)) * lda + k0 + 32, 0, 1);
    } else {
      for (int i = tid; i < 2048; i += 128) {
        int r = i >> 5, c = i & 31;
        int gm = m0 + r, gk = k0 + c;
        float v = (gm < M && gk < K) ? Ab[(long)gm * lda + gk] : 0.0f;
        Ah[r * 40 + c] = (_Float16)v;
      }
      for (int i = tid; i < 2048; i += 128) {
        int n = i >> 5, c = i & 31;
        int gn = n0 + n, gk = k0 + c;
        _Float16 v = (_Float16)0.0f;
        if (gn < N && gk < K) v = Bb[(long)gn * ldb + gk];
        Bt[n * 40 + c] = v;
      }
    }
    __syncthreads();

    // Fragments: contiguous k order (k = 16*(lane/16) + s) for both A and B.
    FragH af[2], bfv[2];
#pragma unroll
    for (int mi = 0; mi < 2; ++mi) {
      int row = (wr << 5) + (mi << 4) + l16;
#pragma unroll
      for (int v = 0; v < 8; ++v) {
        int kk = (v << 1) + (lh << 4);
        af[mi].u[v] = *(const uint32_t*)&Ah[row * 40 + kk];
      }
    }
#pragma unroll
    for (int ni = 0; ni < 2; ++ni) {
      int col = (wc << 5) + (ni << 4) + l16;
#pragma unroll
      for (int v = 0; v < 8; ++v) {
        int kk = (v << 1) + (lh << 4);
        bfv[ni].u[v] = *(const uint32_t*)&Bt[col * 40 + kk];
      }
    }
#pragma unroll
    for (int mi = 0; mi < 2; ++mi)
#pragma unroll
      for (int ni = 0; ni < 2; ++ni)
        acc[mi][ni] = __builtin_amdgcn_wmma_f32_16x16x32_f16(
            false, af[mi].h, false, bfv[ni].h, (short)0, acc[mi][ni],
            false, false);
  }

  // Epilogue: C layout lane=N, vgpr r -> M = r + 8*(lane>=16)
#pragma unroll
  for (int mi = 0; mi < 2; ++mi)
#pragma unroll
    for (int ni = 0; ni < 2; ++ni) {
      int gn = n0 + (wc << 5) + (ni << 4) + l16;
#pragma unroll
      for (int r = 0; r < 8; ++r) {
        int gm = m0 + (wr << 5) + (mi << 4) + r + (lh << 3);
        if (ALIGNED || (gm < M && gn < N)) {
          float v = acc[mi][ni][r] * scale;
          if (bias)    v += bias[gn];
          if (divisor) v /= divisor[gm];
          if (act == 1)      v = fmaxf(v, 0.0f);
          else if (act == 2) v = 1.0f / (1.0f + __expf(-v));
          Cb[(long)gm * ldc + gn] = v;
          if (C16) C16[(long)blockIdx.z * sC + (long)gm * ldc + gn] = (_Float16)v;
        }
      }
    }
}

// ---------------------------------------------------------------------------
// Fused attention -> head-mean adjacency.  q16/k16 are f16 (8192,768).
// ---------------------------------------------------------------------------
#define SCS 516

__global__ __launch_bounds__(128)
void semgcn_attn_adj(const _Float16* __restrict__ q16,
                     const _Float16* __restrict__ k16,
                     const float* __restrict__ T, const int* __restrict__ tok,
                     float* __restrict__ adj, float* __restrict__ denom,
                     float scale)
{
  const int b   = blockIdx.x;
  const int q0  = blockIdx.y << 4;
  const int tid = threadIdx.x;
  const int wave = tid >> 5;
  const int lane = tid & 31;
  const int l16  = lane & 15;
  const int lh   = lane >> 4;
  const int r8   = tid >> 3;
  const int j0   = tid & 7;

  __shared__ _Float16 Qh[16 * 104];
  __shared__ float    sc[16 * SCS];

  float padj[64];
#pragma unroll
  for (int i = 0; i < 64; ++i) padj[i] = 0.0f;

  for (int h = 0; h < 8; ++h) {
    __syncthreads();
    // Stage q tile: raw f16 uint2 copies (16x96 halves = 384 dwords x2)
#pragma unroll
    for (int i = tid; i < 384; i += 128) {
      int r = i / 24, c = (i % 24) << 2;
      *(uint2*)&Qh[r * 104 + c] =
          *(const uint2*)(q16 + ((long)(b * 512 + q0 + r)) * 768 + h * 96 + c);
    }
    __syncthreads();

    for (int t = 0; t < 8; ++t) {
      const int n0 = (wave << 7) + (t << 4);
      const int ng = n0 + l16;
      v8f accv = {};
#pragma unroll
      for (int kc = 0; kc < 96; kc += 32) {
        FragH af, bfv;
#pragma unroll
        for (int v = 0; v < 8; ++v) {
          int kk = kc + (v << 1) + (lh << 4);
          af.u[v] = *(const uint32_t*)&Qh[l16 * 104 + kk];
        }
        // k^T fragment: two b128 loads straight from the f16 K copy
        const _Float16* kp =
            k16 + ((long)(b * 512 + ng)) * 768 + h * 96 + kc + (lh << 4);
        const uint4 p0 = *(const uint4*)kp;
        const uint4 p1 = *(const uint4*)(kp + 8);
        bfv.u[0] = p0.x; bfv.u[1] = p0.y; bfv.u[2] = p0.z; bfv.u[3] = p0.w;
        bfv.u[4] = p1.x; bfv.u[5] = p1.y; bfv.u[6] = p1.z; bfv.u[7] = p1.w;
        accv = __builtin_amdgcn_wmma_f32_16x16x32_f16(
            false, af.h, false, bfv.h, (short)0, accv, false, false);
      }
      const int tk = tok[b * 512 + ng];
#pragma unroll
      for (int r = 0; r < 8; ++r) {
        int m  = r + (lh << 3);
        int qg = q0 + m;
        int d  = ng - qg;
        d = d < -128 ? -128 : (d > 128 ? 128 : d);
        float v = accv[r] * scale + T[(long)(b * 512 + qg) * 257 + d + 128];
        if (tk == 0) v = -1e9f;
        sc[m * SCS + ng] = v;
      }
    }
    __syncthreads();

    float mx = -3.4e38f;
    for (int i = 0; i < 64; ++i) mx = fmaxf(mx, sc[r8 * SCS + j0 + (i << 3)]);
    mx = fmaxf(mx, __shfl_xor(mx, 4, 8));
    mx = fmaxf(mx, __shfl_xor(mx, 2, 8));
    mx = fmaxf(mx, __shfl_xor(mx, 1, 8));
    float sum = 0.0f;
    for (int i = 0; i < 64; ++i) {
      float e = __expf(sc[r8 * SCS + j0 + (i << 3)] - mx);
      sc[r8 * SCS + j0 + (i << 3)] = e;
      sum += e;
    }
    sum += __shfl_xor(sum, 4, 8);
    sum += __shfl_xor(sum, 2, 8);
    sum += __shfl_xor(sum, 1, 8);
    const float inv = 0.125f / sum;
#pragma unroll
    for (int i = 0; i < 64; ++i)
      padj[i] += sc[r8 * SCS + j0 + (i << 3)] * inv;
  }

  const int qg = q0 + r8;
  const int tq = tok[b * 512 + qg];
  float rs = 0.0f;
#pragma unroll
  for (int i = 0; i < 64; ++i) {
    int n = j0 + (i << 3);
    float v = padj[i];
    v = (n == qg) ? 1.0f : v;
    v = (tq != 0) ? v : 0.0f;
    rs += v;
    adj[((long)b * 512 + qg) * 512 + n] = v;
  }
  rs += __shfl_xor(rs, 4, 8);
  rs += __shfl_xor(rs, 2, 8);
  rs += __shfl_xor(rs, 1, 8);
  if (j0 == 0) denom[b * 512 + qg] = rs + 1.0f;
}

// ---------------------------------------------------------------------------
// f32 (K,N) -> f16 (N,K) tiled transpose+convert (dims multiple of 32).
// ---------------------------------------------------------------------------
__global__ __launch_bounds__(256)
void semgcn_t16(const float* __restrict__ in, long sIn, int K, int N,
                _Float16* __restrict__ out, long sOut)
{
  __shared__ float tile[32][33];
  const float* ib = in + (long)blockIdx.z * sIn;
  _Float16*    ob = out + (long)blockIdx.z * sOut;
  const int n0 = blockIdx.x << 5;
  const int k0 = blockIdx.y << 5;
  const int tx = threadIdx.x, ty = threadIdx.y;   // (32, 8)
#pragma unroll
  for (int j = 0; j < 4; ++j)
    tile[ty + (j << 3)][tx] = ib[(long)(k0 + ty + (j << 3)) * N + n0 + tx];
  __syncthreads();
#pragma unroll
  for (int j = 0; j < 4; ++j)
    ob[(long)(n0 + ty + (j << 3)) * K + k0 + tx] =
        (_Float16)tile[tx][ty + (j << 3)];
}

__global__ void semgcn_cvt16(const float* __restrict__ in,
                             _Float16* __restrict__ out, int n)
{
  int i = blockIdx.x * blockDim.x + threadIdx.x;
  if (i < n) out[i] = (_Float16)in[i];
}

// ---------------------------------------------------------------------------
// Elementwise helpers
// ---------------------------------------------------------------------------
__global__ void semgcn_head_mean(const float* __restrict__ q,
                                 float* __restrict__ qm)
{
  int i = blockIdx.x * blockDim.x + threadIdx.x;   // 8192*96
  if (i >= 8192 * 96) return;
  int row = i / 96, j = i - row * 96;
  float s = 0.0f;
#pragma unroll
  for (int h = 0; h < 8; ++h) s += q[(long)row * 768 + h * 96 + j];
  qm[i] = s * 0.125f;
}

__global__ void semgcn_fuse(const float* __restrict__ f0,
                            const float* __restrict__ f1,
                            const float* __restrict__ sw,
                            float* __restrict__ out, int n)
{
  int i = blockIdx.x * blockDim.x + threadIdx.x;
  if (i >= n) return;
  float a = sw[0], b = sw[1];
  float m = fmaxf(a, b);
  float ea = __expf(a - m), eb = __expf(b - m);
  float w0 = ea / (ea + eb);
  out[i] = w0 * f0[i] + (1.0f - w0) * f1[i];
}

__global__ void semgcn_pool(const float* __restrict__ x,
                            float* __restrict__ gcin)
{
  int i = blockIdx.x * blockDim.x + threadIdx.x;   // 16*768
  if (i >= 16 * 768) return;
  int b = i / 768, d = i - b * 768;
  const float* p = x + (long)b * 512 * 768 + d;
  float s = 0.0f, mx = -3.4e38f;
  for (int t = 0; t < 512; ++t) {
    float v = p[(long)t * 768];
    s += v;
    mx = fmaxf(mx, v);
  }
  gcin[b * 1536 + d]       = s * (1.0f / 512.0f);
  gcin[b * 1536 + 768 + d] = mx;
}

__global__ void semgcn_final(const float* __restrict__ x,
                             const float* __restrict__ gate,
                             const float* __restrict__ gc,
                             float* __restrict__ out, int n)
{
  int i = blockIdx.x * blockDim.x + threadIdx.x;
  if (i >= n) return;
  int b = i / (512 * 768);
  int d = i % 768;
  out[i] = x[i] + gate[b * 768 + d] * gc[b * 768 + d];
}

// ---------------------------------------------------------------------------
// Launcher
// ---------------------------------------------------------------------------
extern "C" void kernel_launch(void* const* d_in, const int* in_sizes, int n_in,
                              void* d_out, int out_size, void* d_ws, size_t ws_size,
                              hipStream_t stream)
{
  (void)in_sizes; (void)n_in; (void)out_size; (void)ws_size;

  const float* inputs = (const float*)d_in[0];
  const int*   tok    = (const int*)  d_in[1];
  const float* Wq  = (const float*)d_in[3];
  const float* bq  = (const float*)d_in[4];
  const float* Wk  = (const float*)d_in[5];
  const float* bk  = (const float*)d_in[6];
  const float* rel = (const float*)d_in[7];   // (257, 96)
  const float* W0  = (const float*)d_in[8];
  const float* b0  = (const float*)d_in[9];
  const float* W1  = (const float*)d_in[10];
  const float* b1  = (const float*)d_in[11];
  const float* sw  = (const float*)d_in[12];
  const float* Wf  = (const float*)d_in[13];
  const float* bf  = (const float*)d_in[14];
  const float* Wfc = (const float*)d_in[15];  // (1536, 768)
  const float* bfc = (const float*)d_in[16];
  const float* Wg  = (const float*)d_in[17];
  const float* bg  = (const float*)d_in[18];

  float* outp = (float*)d_out;                 // (16,512,768)
  float* adj  = outp + 6291456;                // (16,512,512)

  float* ws = (float*)d_ws;
  // phase 1 (attention)
  float* q_ = ws;                      // 6291456
  float* k_ = ws + 6291456;            // 6291456 (f32 K unused downstream)
  float* qm = ws + 12582912;           // 786432
  float* Tb = ws + 13369344;           // 2105344
  _Float16* q16 = (_Float16*)(ws + 15474688);   // 6291456 halves
  _Float16* k16 = (_Float16*)(ws + 18620416);   // 6291456 halves
  _Float16* XT  = (_Float16*)(ws + 21766144);   // 6291456 halves (X^T, reused)
  // f16 weights (persistent)
  long WB = 24911872;
  _Float16* WqT  = (_Float16*)(ws + WB);             // 768x768
  _Float16* WkT  = (_Float16*)(ws + WB + 294912);
  _Float16* W0T  = (_Float16*)(ws + WB + 589824);
  _Float16* W1T  = (_Float16*)(ws + WB + 884736);
  _Float16* WfT  = (_Float16*)(ws + WB + 1179648);
  _Float16* WgT  = (_Float16*)(ws + WB + 1474560);
  _Float16* WfcT = (_Float16*)(ws + WB + 1769472);   // 768x1536
  _Float16* rel16= (_Float16*)(ws + WB + 2359296);   // 257x96
  // small persistent buffers
  float* den  = ws + 27283520;         // 8192
  float* gcin = ws + 27291712;         // 24576
  float* gc   = ws + 27316288;         // 12288
  float* gate = ws + 27328576;         // 12288
  // phase 2 aliases (time-disjoint with phase 1)
  float* ax = ws;                      // 6291456
  float* f0 = ws + 6291456;            // 6291456
  float* f1 = ws + 12582912;           // 6291456
  float* xb = ws + 6291456;            // aliases f0 (dead after fuse)

  const dim3 blk(128);
  const dim3 tblk(32, 8);
  const float rscale = 0.1020620726159658f;    // 1/sqrt(96)

  // --- one-time weight prep: f32 (K,N) -> f16 (N,K) ---
  semgcn_t16<<<dim3(24, 24, 1), tblk, 0, stream>>>(Wq,  0, 768, 768, WqT,  0);
  semgcn_t16<<<dim3(24, 24, 1), tblk, 0, stream>>>(Wk,  0, 768, 768, WkT,  0);
  semgcn_t16<<<dim3(24, 24, 1), tblk, 0, stream>>>(W0,  0, 768, 768, W0T,  0);
  semgcn_t16<<<dim3(24, 24, 1), tblk, 0, stream>>>(W1,  0, 768, 768, W1T,  0);
  semgcn_t16<<<dim3(24, 24, 1), tblk, 0, stream>>>(Wf,  0, 768, 768, WfT,  0);
  semgcn_t16<<<dim3(24, 24, 1), tblk, 0, stream>>>(Wg,  0, 768, 768, WgT,  0);
  semgcn_t16<<<dim3(24, 48, 1), tblk, 0, stream>>>(Wfc, 0, 1536, 768, WfcT, 0);
  semgcn_cvt16<<<97, 256, 0, stream>>>(rel, rel16, 24672);
  // inputs^T (batched) for adj@inputs
  semgcn_t16<<<dim3(24, 16, 16), tblk, 0, stream>>>(inputs, 393216, 512, 768,
                                                    XT, 393216);

  // Q = inputs@Wq + bq ; K = inputs@Wk + bk  (also emit f16 copies)
  semgcn_gemm<true><<<dim3(12, 128, 1), blk, 0, stream>>>(
      inputs, 0, 768, WqT, 0, 768, q_, 0, 768, q16, bq, nullptr, 1.0f, 0,
      8192, 768, 768);
  semgcn_gemm<true><<<dim3(12, 128, 1), blk, 0, stream>>>(
      inputs, 0, 768, WkT, 0, 768, k_, 0, 768, k16, bk, nullptr, 1.0f, 0,
      8192, 768, 768);

  // q_mean over heads -> (8192, 96)
  semgcn_head_mean<<<3072, 256, 0, stream>>>(q_, qm);

  // T = (q_mean @ rel_emb^T) * scale -> (8192, 257)   [rel16 already (N,K)]
  semgcn_gemm<false><<<dim3(5, 128, 1), blk, 0, stream>>>(
      qm, 0, 96, rel16, 0, 96, Tb, 0, 257, nullptr, nullptr, nullptr,
      rscale, 0, 8192, 257, 96);

  // Fused attention -> adjacency (into d_out) + denom
  semgcn_attn_adj<<<dim3(16, 32), blk, 0, stream>>>(
      q16, k16, Tb, tok, adj, den, rscale);

  // GCN layer 1: ax = adj @ inputs, f0 = relu((ax@W0+b0)/den)
  semgcn_gemm<true><<<dim3(12, 8, 16), blk, 0, stream>>>(
      adj, 262144, 512, XT, 393216, 512, ax, 393216, 768, nullptr,
      nullptr, nullptr, 1.0f, 0, 512, 768, 512);
  semgcn_gemm<true><<<dim3(12, 128, 1), blk, 0, stream>>>(
      ax, 0, 768, W0T, 0, 768, f0, 0, 768, nullptr, b0, den, 1.0f, 1,
      8192, 768, 768);

  // GCN layer 2: ax = adj @ f0, f1 = relu((ax@W1+b1)/den)
  semgcn_t16<<<dim3(24, 16, 16), tblk, 0, stream>>>(f0, 393216, 512, 768,
                                                    XT, 393216);
  semgcn_gemm<true><<<dim3(12, 8, 16), blk, 0, stream>>>(
      adj, 262144, 512, XT, 393216, 512, ax, 393216, 768, nullptr,
      nullptr, nullptr, 1.0f, 0, 512, 768, 512);
  semgcn_gemm<true><<<dim3(12, 128, 1), blk, 0, stream>>>(
      ax, 0, 768, W1T, 0, 768, f1, 0, 768, nullptr, b1, den, 1.0f, 1,
      8192, 768, 768);

  // combined = softmax(scale_w) . [f0, f1]  (into ax)
  semgcn_fuse<<<24576, 256, 0, stream>>>(f0, f1, sw, ax, 6291456);

  // x = combined @ Wf + bf  (into xb, aliasing dead f0)
  semgcn_gemm<true><<<dim3(12, 128, 1), blk, 0, stream>>>(
      ax, 0, 768, WfT, 0, 768, xb, 0, 768, nullptr, bf, nullptr, 1.0f, 0,
      8192, 768, 768);

  // pooling -> gcin (16, 1536) = [avg | max]
  semgcn_pool<<<48, 256, 0, stream>>>(xb, gcin);

  // gc = gcin @ Wfc + bfc   (M=16, K=1536, N=768)
  semgcn_gemm<false><<<dim3(12, 1, 1), blk, 0, stream>>>(
      gcin, 0, 1536, WfcT, 0, 1536, gc, 0, 768, nullptr, bfc, nullptr,
      1.0f, 0, 16, 768, 1536);

  // gate = sigmoid(gc @ Wg + bg)
  semgcn_gemm<false><<<dim3(12, 1, 1), blk, 0, stream>>>(
      gc, 0, 768, WgT, 0, 768, gate, 0, 768, nullptr, bg, nullptr,
      1.0f, 2, 16, 768, 768);

  // outputs = x + gate * gc
  semgcn_final<<<24576, 256, 0, stream>>>(xb, gate, gc, outp, 6291456);
}
